// VishwamAITransformerLayer_31086973288505
// MI455X (gfx1250) — matrix operand order repaired and checked by reading, not compiled, not run on hardware
//
#include <hip/hip_runtime.h>

typedef __bf16 bhalf;
typedef __attribute__((ext_vector_type(16))) __bf16 v16bf;
typedef __attribute__((ext_vector_type(8)))  __bf16 v8bf;
typedef __attribute__((ext_vector_type(8)))  float  v8f;

constexpr int CB = 2, CS = 1024, CD = 1024, CH = 16, CDH = 64, CFF = 4096;
constexpr int CE = 4, CKK = 64, CMF = 256, CG = 128;
constexpr int CT = CB * CS;      // 2048 tokens
constexpr int CBH = CB * CH;     // 32 (b,h) batches
constexpr float F_DEPTH = 1.25f;     // 1 + 12/24*0.5
constexpr float F_SCALE = 0.125f;    // 1/sqrt(64)

// ---------------- workspace arena ----------------
static constexpr size_t AL2(size_t x) { return (x + 255) & ~(size_t)255; }
constexpr size_t TDb2 = (size_t)CT * CD * 2, TDb4 = (size_t)CT * CD * 4;
constexpr size_t SZw  = AL2((size_t)CD * CD * 2);
constexpr size_t O_xb   = 0;
constexpr size_t O_moeT = AL2(TDb2);                 // 16 x (DxD) bf16 N-major
constexpr size_t O_spT  = O_moeT + 16 * SZw;
constexpr size_t O_pfT  = O_spT + 4 * SZw;
constexpr size_t O_wfT  = O_pfT + 4 * SZw;           // MF x DH
constexpr size_t O_f1T  = O_wfT + AL2((size_t)CMF * CDH * 2);
constexpr size_t O_f2T  = O_f1T + AL2((size_t)CFF * CD * 2);
constexpr size_t O_Qb   = O_f2T + AL2((size_t)CD * CFF * 2);
constexpr size_t O_Kb   = O_Qb + AL2(TDb2);
constexpr size_t O_Vb   = O_Kb + AL2(TDb2);
constexpr size_t O_Vt   = O_Vb + AL2(TDb2);          // (B*H) x DH x S bf16
constexpr size_t O_AOb  = O_Vt + AL2((size_t)CBH * CDH * CS * 2);
constexpr size_t O_x1b  = O_AOb + AL2(TDb2);
constexpr size_t O_EO   = O_x1b + AL2(TDb2);         // f32 from here on
constexpr size_t O_moeO = O_EO + AL2(TDb4);
constexpr size_t O_spO  = O_moeO + AL2(TDb4);
constexpr size_t O_pfO  = O_spO + AL2(TDb4);
constexpr size_t O_num  = O_pfO + AL2(TDb4);
constexpr size_t O_y    = O_num + AL2(TDb4);
constexpr size_t O_x1   = O_y + AL2(TDb4);
constexpr size_t O_ffn  = O_x1 + AL2(TDb4);
constexpr size_t O_y2   = O_ffn + AL2(TDb4);
constexpr size_t O_x2   = O_y2 + AL2(TDb4);
constexpr size_t O_avg  = O_x2 + AL2(TDb4);
constexpr size_t O_gw   = O_avg + AL2((size_t)CB * CD * 4);
constexpr size_t O_tg   = O_gw + 256;
constexpr size_t O_ag   = O_tg + 256;
constexpr size_t O_rg   = O_ag + 256;
constexpr size_t O_z    = O_rg + AL2((size_t)CT * CE * 4);
constexpr size_t O_thr  = O_z + AL2((size_t)CBH * CMF * 4);
constexpr size_t O_SCR  = O_thr + AL2((size_t)CBH * CS * 4);
// scratch aliases (phases are stream-ordered)
constexpr size_t O_scores = O_SCR;                                    // 128MB f32
constexpr size_t O_probs  = O_SCR + AL2((size_t)CBH * CS * CS * 4);   // 64MB bf16
constexpr size_t O_pq  = O_SCR;
constexpr size_t O_pk  = O_pq + AL2((size_t)CBH * CS * CMF * 2);
constexpr size_t O_pkT = O_pk + AL2((size_t)CBH * CS * CMF * 2);
constexpr size_t O_kvT = O_pkT + AL2((size_t)CBH * CS * CMF * 2);
constexpr size_t O_h1  = O_SCR;                                       // T x FF bf16

// ---------------- device helpers ----------------
__device__ __forceinline__ float wred_sum(float v) {
  for (int m = 16; m; m >>= 1) v += __shfl_xor(v, m, 32);
  return v;
}
__device__ __forceinline__ float wred_max(float v) {
  for (int m = 16; m; m >>= 1) v = fmaxf(v, __shfl_xor(v, m, 32));
  return v;
}
__device__ __forceinline__ float gelu_t(float x) {
  float x3 = x * x * x;
  return 0.5f * x * (1.f + tanhf(0.7978845608028654f * (x + 0.044715f * x3)));
}
__device__ __forceinline__ float sigm(float x) { return 1.f / (1.f + __expf(-x)); }
__device__ __forceinline__ unsigned fkey(float f) {
  unsigned u = __float_as_uint(f);
  return (u & 0x80000000u) ? ~u : (u | 0x80000000u);
}
__device__ __forceinline__ float kinv(unsigned k) {
  unsigned u = (k & 0x80000000u) ? (k ^ 0x80000000u) : ~k;
  return __uint_as_float(u);
}

// ---------------- WMMA GEMM: C = act(scale*(A @ Bt^T) + bias) ----------------
// A: M x K bf16 row-major (lda), Bt: N x K bf16 row-major (ldb).
// Block = 128 threads (4 waves), block tile 64x64, wave tile 32x32 (2x2 WMMA),
// K step 32. batch z = blockIdx.z split as outer = z/inner, innerIdx = z%inner.
__global__ __launch_bounds__(128) void k_gemm(
    const bhalf* __restrict__ A, const bhalf* __restrict__ Bt,
    const float* __restrict__ bias, float* __restrict__ C, bhalf* __restrict__ Cb,
    int K, int lda, int ldb, int ldc, int inner,
    long sAo, long sAi, long sBo, long sBi, long sCo, long sCi,
    float scale, int act)
{
  int z = blockIdx.z;
  int zo = z / inner, zi = z % inner;
  A  += (long)zo * sAo + (long)zi * sAi;
  Bt += (long)zo * sBo + (long)zi * sBi;
  long cbase = (long)zo * sCo + (long)zi * sCi;

  __shared__ __align__(16) bhalf As[64 * 32];
  __shared__ __align__(16) bhalf Bs[64 * 32];

  int tid = threadIdx.x, lane = tid & 31, wave = tid >> 5;
  int wm = (wave & 1) * 32, wn = (wave >> 1) * 32;
  int m0 = blockIdx.x * 64, n0 = blockIdx.y * 64;

  int lr = tid >> 1;                 // tile row 0..63 for staging loads
  int lc = (tid & 1) * 16;           // col half 0 / 16

  v8f acc[2][2];
  v8f vz = {0.f, 0.f, 0.f, 0.f, 0.f, 0.f, 0.f, 0.f};
  acc[0][0] = vz; acc[0][1] = vz; acc[1][0] = vz; acc[1][1] = vz;

  int half = lane >> 4, l16 = lane & 15;

  for (int k0 = 0; k0 < K; k0 += 32) {
    const bhalf* ga = A + (long)(m0 + lr) * lda + k0 + lc;
    const bhalf* gb = Bt + (long)(n0 + lr) * ldb + k0 + lc;
    *(v8bf*)&As[lr * 32 + lc]     = *(const v8bf*)ga;
    *(v8bf*)&As[lr * 32 + lc + 8] = *(const v8bf*)(ga + 8);
    *(v8bf*)&Bs[lr * 32 + lc]     = *(const v8bf*)gb;
    *(v8bf*)&Bs[lr * 32 + lc + 8] = *(const v8bf*)(gb + 8);
    if (k0 + 32 < K) {                       // pull next K-tile toward L2
      __builtin_prefetch(ga + 32, 0, 1);
      __builtin_prefetch(gb + 32, 0, 1);
    }
    __syncthreads();

    v16bf af[2], bfr[2];
    #pragma unroll
    for (int mi = 0; mi < 2; ++mi) {
      // A frag: lanes 0-15 hold K {0..7,16..23}; lanes 16-31 hold K {8..15,24..31}
      const bhalf* p = &As[(wm + mi * 16 + l16) * 32 + half * 8];
      v8bf lo = *(const v8bf*)p;
      v8bf hi = *(const v8bf*)(p + 16);
      #pragma unroll
      for (int i = 0; i < 8; ++i) { af[mi][i] = lo[i]; af[mi][i + 8] = hi[i]; }
    }
    #pragma unroll
    for (int ni = 0; ni < 2; ++ni) {
      // B frag: lanes 0-15 hold K {0..7,16..23} of col l16; lanes 16-31 hold K {8..15,24..31}
      const bhalf* p = &Bs[(wn + ni * 16 + l16) * 32 + half * 8];
      v8bf lo = *(const v8bf*)p;
      v8bf hi = *(const v8bf*)(p + 16);
      #pragma unroll
      for (int i = 0; i < 8; ++i) { bfr[ni][i] = lo[i]; bfr[ni][i + 8] = hi[i]; }
    }
    #pragma unroll
    for (int mi = 0; mi < 2; ++mi)
      #pragma unroll
      for (int ni = 0; ni < 2; ++ni)
        acc[mi][ni] = __builtin_amdgcn_wmma_f32_16x16x32_bf16(
            false, af[mi], false, bfr[ni], (short)0, acc[mi][ni], false, false);
    __syncthreads();
  }

  // epilogue: C/D layout -> lane<16: M=v, lane>=16: M=8+v, N=lane&15
  #pragma unroll
  for (int mi = 0; mi < 2; ++mi)
    #pragma unroll
    for (int ni = 0; ni < 2; ++ni) {
      int colg = n0 + wn + ni * 16 + l16;
      float bv = bias ? bias[colg] : 0.f;
      #pragma unroll
      for (int v = 0; v < 8; ++v) {
        int rowg = m0 + wm + mi * 16 + half * 8 + v;
        float r = acc[mi][ni][v] * scale + bv;
        if (act == 1) r = gelu_t(r);
        else if (act == 2) r = fmaxf(r, 0.f);
        long off = cbase + (long)rowg * ldc + colg;
        if (C)  C[off]  = r;
        if (Cb) Cb[off] = (bhalf)r;
      }
    }
}

// ---------------- small kernels ----------------
__global__ void k_cvt_bf16(const float* __restrict__ in, bhalf* __restrict__ out, long n) {
  long i = (long)blockIdx.x * blockDim.x + threadIdx.x;
  if (i < n) out[i] = (bhalf)in[i];
}
__global__ void k_fill(float* p, float v, long n) {
  long i = (long)blockIdx.x * blockDim.x + threadIdx.x;
  if (i < n) p[i] = v;
}
// bt[n*Kd + k] = (bf16) w[k*Nd + n]   (w is Kd x Nd row-major)
__global__ void k_cvtT(const float* __restrict__ w, bhalf* __restrict__ bt, int Kd, int Nd) {
  long i = (long)blockIdx.x * blockDim.x + threadIdx.x;
  if (i < (long)Kd * Nd) {
    int n = (int)(i / Kd), k = (int)(i % Kd);
    bt[i] = (bhalf)w[(long)k * Nd + n];
  }
}
// Vt[(b*H+h)*DH*S + d*S + s] = X[(b*S+s)*D + h*DH + d]
__global__ void k_transpose_bhead(const bhalf* __restrict__ X, bhalf* __restrict__ Out) {
  long i = (long)blockIdx.x * blockDim.x + threadIdx.x;
  if (i >= (long)CBH * CDH * CS) return;
  int s = (int)(i % CS); long r = i / CS;
  int d = (int)(r % CDH); long q = r / CDH;
  int h = (int)(q % CH);  int b = (int)(q / CH);
  Out[i] = X[((long)b * CS + s) * CD + h * CDH + d];
}
// out[z][c*rows + r] = in[z][r*cols + c]
__global__ void k_transpose_b(const bhalf* __restrict__ in, bhalf* __restrict__ out,
                              int rows, int cols, long n) {
  long i = (long)blockIdx.x * blockDim.x + threadIdx.x;
  if (i >= n) return;
  long rc = (long)rows * cols;
  long z = i / rc, rem = i - z * rc;
  int c = (int)(rem / rows), r = (int)(rem % rows);
  out[i] = in[z * rc + (long)r * cols + c];
}
__global__ void k_mean(const float* __restrict__ x, float* __restrict__ avg) {
  int i = blockIdx.x * blockDim.x + threadIdx.x;
  if (i >= CB * CD) return;
  int b = i / CD, d = i % CD;
  float s = 0.f;
  for (int ss = 0; ss < CS; ++ss) s += x[((long)b * CS + ss) * CD + d];
  avg[i] = s / (float)CS;
}
__global__ __launch_bounds__(CG) void k_gates(
    const float* __restrict__ avg,
    const float* w1, const float* b1, const float* w2, const float* b2,
    const float* tw, const float* tb, const float* aw, const float* ab,
    float* gw, float* tg, float* ag)
{
  int b = blockIdx.x, n = threadIdx.x;
  const float* a = avg + (long)b * CD;
  __shared__ float hs[CG];
  __shared__ float tv[2], av[2];
  float acc = b1[n];
  for (int d = 0; d < CD; ++d) acc += a[d] * w1[(long)d * CG + n];
  hs[n] = gelu_t(acc);
  __syncthreads();
  if (n < 2) {
    float o = b2[n];
    for (int k = 0; k < CG; ++k) o += hs[k] * w2[k * 2 + n];
    gw[b * 2 + n] = sigm(o);
    float t = tb[n], g = ab[n];
    for (int d = 0; d < CD; ++d) { t += a[d] * tw[d * 2 + n]; g += a[d] * aw[d * 2 + n]; }
    tv[n] = t; av[n] = g;
  }
  __syncthreads();
  if (n == 0) {
    float m = fmaxf(tv[0], tv[1]);
    float e0 = __expf(tv[0] - m), e1 = __expf(tv[1] - m);
    tg[b * 2] = e0 / (e0 + e1); tg[b * 2 + 1] = e1 / (e0 + e1);
    m = fmaxf(av[0], av[1]);
    e0 = __expf(av[0] - m); e1 = __expf(av[1] - m);
    ag[b * 2] = e0 / (e0 + e1); ag[b * 2 + 1] = e1 / (e0 + e1);
  }
}
__global__ __launch_bounds__(256) void k_router(const float* __restrict__ x,
                                                const float* wr, const float* br,
                                                float* __restrict__ g) {
  int wave = threadIdx.x >> 5, lane = threadIdx.x & 31;
  long t = (long)blockIdx.x * 8 + wave;
  if (t >= CT) return;
  float acc[CE] = {0.f, 0.f, 0.f, 0.f};
  for (int d = lane; d < CD; d += 32) {
    float xv = x[t * CD + d];
    #pragma unroll
    for (int e = 0; e < CE; ++e) acc[e] += xv * wr[(long)d * CE + e];
  }
  #pragma unroll
  for (int e = 0; e < CE; ++e) acc[e] = wred_sum(acc[e]) + br[e];
  if (lane == 0) {
    float m = fmaxf(fmaxf(acc[0], acc[1]), fmaxf(acc[2], acc[3]));
    float s = 0.f, ev[CE];
    #pragma unroll
    for (int e = 0; e < CE; ++e) { ev[e] = __expf(acc[e] - m); s += ev[e]; }
    #pragma unroll
    for (int e = 0; e < CE; ++e) g[t * CE + e] = ev[e] / s;
  }
}
__global__ __launch_bounds__(256) void k_topk(const float* __restrict__ Sc,
                                              float* __restrict__ thr) {
  int wave = threadIdx.x >> 5, lane = threadIdx.x & 31;
  long row = (long)blockIdx.x * 8 + wave;
  const float* r = Sc + row * CS;
  unsigned keys[32];
  #pragma unroll
  for (int i = 0; i < 32; ++i) keys[i] = fkey(r[lane + 32 * i]);
  unsigned t = 0;
  for (int bit = 31; bit >= 0; --bit) {      // max t with count(keys>=t) >= K
    unsigned cand = t | (1u << bit);
    int c = 0;
    #pragma unroll
    for (int i = 0; i < 32; ++i) c += (keys[i] >= cand) ? 1 : 0;
    for (int m = 16; m; m >>= 1) c += __shfl_xor(c, m, 32);
    if (c >= CKK) t = cand;
  }
  if (lane == 0) thr[row] = kinv(t);
}
__global__ __launch_bounds__(256) void k_softmax(const float* __restrict__ Sc,
                                                 const float* __restrict__ thr,
                                                 bhalf* __restrict__ P) {
  int wave = threadIdx.x >> 5, lane = threadIdx.x & 31;
  long row = (long)blockIdx.x * 8 + wave;
  const float* r = Sc + row * CS;
  float th = thr ? thr[row] : -3.0e38f;
  float v[32], mx = -3.0e38f;
  #pragma unroll
  for (int i = 0; i < 32; ++i) {
    float a = r[lane + 32 * i];
    if (a < th) a = -1.0e9f;
    v[i] = a; mx = fmaxf(mx, a);
  }
  mx = wred_max(mx);
  float s = 0.f;
  #pragma unroll
  for (int i = 0; i < 32; ++i) { v[i] = __expf(v[i] - mx); s += v[i]; }
  s = wred_sum(s);
  float inv = 1.f / s;
  bhalf* o = P + row * CS;
  #pragma unroll
  for (int i = 0; i < 32; ++i) o[lane + 32 * i] = (bhalf)(v[i] * inv);
}
__global__ __launch_bounds__(256) void k_layernorm(const float* __restrict__ X,
                                                   const float* g, const float* be,
                                                   float* __restrict__ Y,
                                                   bhalf* __restrict__ Yb) {
  int wave = threadIdx.x >> 5, lane = threadIdx.x & 31;
  long row = (long)blockIdx.x * 8 + wave;
  const float* r = X + row * CD;
  float v[32], s1 = 0.f, s2 = 0.f;
  #pragma unroll
  for (int i = 0; i < 32; ++i) { v[i] = r[lane + 32 * i]; s1 += v[i]; s2 += v[i] * v[i]; }
  s1 = wred_sum(s1); s2 = wred_sum(s2);
  float mu = s1 / (float)CD;
  float var = s2 / (float)CD - mu * mu;
  float rs = rsqrtf(var + 1e-5f);
  #pragma unroll
  for (int i = 0; i < 32; ++i) {
    int d = lane + 32 * i;
    float o = (v[i] - mu) * rs * g[d] + be[d];
    Y[row * CD + d] = o;
    if (Yb) Yb[row * CD + d] = (bhalf)o;
  }
}
__global__ void k_moe_accum(const float* __restrict__ eo, const float* __restrict__ g,
                            float* __restrict__ out, int e) {
  long i = (long)blockIdx.x * blockDim.x + threadIdx.x;
  if (i >= (long)CT * CD) return;
  long t = i / CD;
  out[i] += g[t * CE + e] * eo[i];
}
__global__ void k_colsum(const bhalf* __restrict__ pk, float* __restrict__ z) {
  int i = blockIdx.x * blockDim.x + threadIdx.x;
  if (i >= CBH * CMF) return;
  int zb = i / CMF, m = i % CMF;
  const bhalf* p = pk + (long)zb * CS * CMF + m;
  float s = 0.f;
  for (int ss = 0; ss < CS; ++ss) s += (float)p[(long)ss * CMF];
  z[i] = s;
}
__global__ __launch_bounds__(256) void k_pffin(const float* __restrict__ num,
                                               const bhalf* __restrict__ pq,
                                               const float* __restrict__ z,
                                               bhalf* __restrict__ outb) {
  int wave = threadIdx.x >> 5, lane = threadIdx.x & 31;
  long r = (long)blockIdx.x * 8 + wave;      // over B*H*S
  int s = (int)(r % CS); long q = r / CS;
  int h = (int)(q % CH); int b = (int)(q / CH);
  const bhalf* pr = pq + ((long)(b * CH + h) * CS + s) * CMF;
  const float* zr = z + (long)(b * CH + h) * CMF;
  float d = 0.f;
  for (int m = lane; m < CMF; m += 32) d += (float)pr[m] * zr[m];
  d = wred_sum(d) + 1e-6f;
  float inv = 1.f / d;
  long base = ((long)b * CS + s) * CD + h * CDH;
  for (int dd = lane; dd < CDH; dd += 32) outb[base + dd] = (bhalf)(num[base + dd] * inv);
}
__global__ void k_combine_attn(const float* __restrict__ x, const float* __restrict__ moe,
                               const float* __restrict__ sp, const float* __restrict__ pf,
                               const float* tg, const float* ag, float* __restrict__ y) {
  long i = (long)blockIdx.x * blockDim.x + threadIdx.x;
  if (i >= (long)CT * CD) return;
  int b = (int)(i / ((long)CS * CD));
  float taa = tg[b * 2] * sp[i] + tg[b * 2 + 1] * pf[i];
  y[i] = x[i] + F_DEPTH * (ag[b * 2] * moe[i] + ag[b * 2 + 1] * taa);
}
__global__ void k_combine_ffn(const float* __restrict__ x1, const float* __restrict__ f,
                              const float* gw, float* __restrict__ y) {
  long i = (long)blockIdx.x * blockDim.x + threadIdx.x;
  if (i >= (long)CT * CD) return;
  int b = (int)(i / ((long)CS * CD));
  y[i] = x1[i] + f[i] * (gw[b * 2 + 1] * F_DEPTH);
}

// ---------------- host ----------------
extern "C" void kernel_launch(void* const* d_in, const int* in_sizes, int n_in,
                              void* d_out, int out_size, void* d_ws, size_t ws_size,
                              hipStream_t stream) {
  (void)in_sizes; (void)n_in; (void)out_size; (void)ws_size;
  const float* x      = (const float*)d_in[0];
  const float* arg_w1 = (const float*)d_in[1];
  const float* arg_b1 = (const float*)d_in[2];
  const float* arg_w2 = (const float*)d_in[3];
  const float* arg_b2 = (const float*)d_in[4];
  const float* moe_wr = (const float*)d_in[5];
  const float* moe_br = (const float*)d_in[6];
  const float* moe_w[4] = {(const float*)d_in[7], (const float*)d_in[9],
                           (const float*)d_in[11], (const float*)d_in[13]};
  const float* moe_b[4] = {(const float*)d_in[8], (const float*)d_in[10],
                           (const float*)d_in[12], (const float*)d_in[14]};
  const float* sp_w[4] = {(const float*)d_in[15], (const float*)d_in[17],
                          (const float*)d_in[19], (const float*)d_in[21]};
  const float* sp_b[4] = {(const float*)d_in[16], (const float*)d_in[18],
                          (const float*)d_in[20], (const float*)d_in[22]};
  const float* pf_w[4] = {(const float*)d_in[23], (const float*)d_in[25],
                          (const float*)d_in[27], (const float*)d_in[29]};
  const float* pf_b[4] = {(const float*)d_in[24], (const float*)d_in[26],
                          (const float*)d_in[28], (const float*)d_in[30]};
  const float* pf_wf  = (const float*)d_in[31];
  const float* taa_wg = (const float*)d_in[32];
  const float* taa_bg = (const float*)d_in[33];
  const float* ag_w   = (const float*)d_in[34];
  const float* ag_b   = (const float*)d_in[35];
  const float* ffn_w1 = (const float*)d_in[36];
  const float* ffn_b1 = (const float*)d_in[37];
  const float* ffn_w2 = (const float*)d_in[38];
  const float* ffn_b2 = (const float*)d_in[39];
  const float* n1_g = (const float*)d_in[40]; const float* n1_b = (const float*)d_in[41];
  const float* n2_g = (const float*)d_in[42]; const float* n2_b = (const float*)d_in[43];
  const float* n3_g = (const float*)d_in[44]; const float* n3_b = (const float*)d_in[45];
  float* out = (float*)d_out;
  char* ws = (char*)d_ws;

  bhalf* xb   = (bhalf*)(ws + O_xb);
  bhalf* Qb   = (bhalf*)(ws + O_Qb);
  bhalf* Kb   = (bhalf*)(ws + O_Kb);
  bhalf* Vb   = (bhalf*)(ws + O_Vb);
  bhalf* Vt   = (bhalf*)(ws + O_Vt);
  bhalf* AOb  = (bhalf*)(ws + O_AOb);
  bhalf* x1b  = (bhalf*)(ws + O_x1b);
  bhalf* wfT  = (bhalf*)(ws + O_wfT);
  bhalf* f1T  = (bhalf*)(ws + O_f1T);
  bhalf* f2T  = (bhalf*)(ws + O_f2T);
  bhalf* probsB = (bhalf*)(ws + O_probs);
  bhalf* pqB  = (bhalf*)(ws + O_pq);
  bhalf* pkB  = (bhalf*)(ws + O_pk);
  bhalf* pkTB = (bhalf*)(ws + O_pkT);
  bhalf* kvTB = (bhalf*)(ws + O_kvT);
  bhalf* h1B  = (bhalf*)(ws + O_h1);
  float* scoresF = (float*)(ws + O_scores);
  float* EO   = (float*)(ws + O_EO);
  float* moeO = (float*)(ws + O_moeO);
  float* spO  = (float*)(ws + O_spO);
  float* pfO  = (float*)(ws + O_pfO);
  float* numF = (float*)(ws + O_num);
  float* yF   = (float*)(ws + O_y);
  float* x1F  = (float*)(ws + O_x1);
  float* ffnF = (float*)(ws + O_ffn);
  float* y2F  = (float*)(ws + O_y2);
  float* x2F  = (float*)(ws + O_x2);
  float* avgF = (float*)(ws + O_avg);
  float* gwF  = (float*)(ws + O_gw);
  float* tgF  = (float*)(ws + O_tg);
  float* agF  = (float*)(ws + O_ag);
  float* rgF  = (float*)(ws + O_rg);
  float* zF   = (float*)(ws + O_z);
  float* thrF = (float*)(ws + O_thr);

  auto gemm = [&](const bhalf* A, const bhalf* Bt, const float* bias,
                  float* C, bhalf* Cb, int M, int N, int K,
                  int lda, int ldb, int ldc, int nb, int inner,
                  long sAo, long sAi, long sBo, long sBi, long sCo, long sCi,
                  float scale, int act) {
    dim3 g(M / 64, N / 64, nb);
    k_gemm<<<g, 128, 0, stream>>>(A, Bt, bias, C, Cb, K, lda, ldb, ldc, inner,
                                  sAo, sAi, sBo, sBi, sCo, sCi, scale, act);
  };
  auto cvtT = [&](const float* w, size_t off, int Kd, int Nd) {
    long n = (long)Kd * Nd;
    k_cvtT<<<(unsigned)((n + 255) / 256), 256, 0, stream>>>(w, (bhalf*)(ws + off), Kd, Nd);
  };

  // 0) activations + weights to bf16 (weights transposed to N x K)
  k_cvt_bf16<<<(unsigned)(((long)CT * CD + 255) / 256), 256, 0, stream>>>(x, xb, (long)CT * CD);
  for (int t = 0; t < 4; ++t)
    for (int e = 0; e < CE; ++e)
      cvtT(moe_w[t] + (size_t)e * CD * CD, O_moeT + (size_t)(t * 4 + e) * SZw, CD, CD);
  for (int t = 0; t < 4; ++t) cvtT(sp_w[t], O_spT + (size_t)t * SZw, CD, CD);
  for (int t = 0; t < 4; ++t) cvtT(pf_w[t], O_pfT + (size_t)t * SZw, CD, CD);
  cvtT(pf_wf, O_wfT, CDH, CMF);
  cvtT(ffn_w1, O_f1T, CD, CFF);
  cvtT(ffn_w2, O_f2T, CFF, CD);

  // 1) gates
  k_mean<<<(CB * CD + 255) / 256, 256, 0, stream>>>(x, avgF);
  k_gates<<<CB, CG, 0, stream>>>(avgF, arg_w1, arg_b1, arg_w2, arg_b2,
                                 taa_wg, taa_bg, ag_w, ag_b, gwF, tgF, agF);
  k_router<<<CT / 8, 256, 0, stream>>>(x, moe_wr, moe_br, rgF);
  k_fill<<<(unsigned)(((long)CT * CD + 255) / 256), 256, 0, stream>>>(moeO, 0.f, (long)CT * CD);

  long eg = (long)CT * CD;  // elementwise grid size
  auto mha = [&](const bhalf* wqT, const float* bq, const bhalf* wkT, const float* bk,
                 const bhalf* wvT, const float* bv, const bhalf* woT, const float* bo,
                 float* outF, bool sparse) {
    gemm(xb, wqT, bq, nullptr, Qb, CT, CD, CD, CD, CD, CD, 1, 1, 0, 0, 0, 0, 0, 0, 1.f, 0);
    gemm(xb, wkT, bk, nullptr, Kb, CT, CD, CD, CD, CD, CD, 1, 1, 0, 0, 0, 0, 0, 0, 1.f, 0);
    gemm(xb, wvT, bv, nullptr, Vb, CT, CD, CD, CD, CD, CD, 1, 1, 0, 0, 0, 0, 0, 0, 1.f, 0);
    long nvt = (long)CBH * CDH * CS;
    k_transpose_bhead<<<(unsigned)((nvt + 255) / 256), 256, 0, stream>>>(Vb, Vt);
    // scores = scale * Q K^T   (per (b,h): A base b*S*D + h*DH)
    gemm(Qb, Kb, nullptr, scoresF, nullptr, CS, CS, CDH, CD, CD, CS, CBH, CH,
         (long)CS * CD, (long)CDH, (long)CS * CD, (long)CDH,
         (long)CH * CS * CS, (long)CS * CS, F_SCALE, 0);
    if (sparse) k_topk<<<CBH * CS / 8, 256, 0, stream>>>(scoresF, thrF);
    k_softmax<<<CBH * CS / 8, 256, 0, stream>>>(scoresF, sparse ? thrF : nullptr, probsB);
    // AO(merged bf16) = P V
    gemm(probsB, Vt, nullptr, nullptr, AOb, CS, CDH, CS, CS, CS, CD, CBH, CH,
         (long)CH * CS * CS, (long)CS * CS, (long)CH * CDH * CS, (long)CDH * CS,
         (long)CS * CD, (long)CDH, 1.f, 0);
    gemm(AOb, woT, bo, outF, nullptr, CT, CD, CD, CD, CD, CD, 1, 1, 0, 0, 0, 0, 0, 0, 1.f, 0);
  };

  // 2) MoE: experts sequentially, weighted accumulate
  for (int e = 0; e < CE; ++e) {
    mha((bhalf*)(ws + O_moeT + (size_t)(0 * 4 + e) * SZw), moe_b[0] + (size_t)e * CD,
        (bhalf*)(ws + O_moeT + (size_t)(1 * 4 + e) * SZw), moe_b[1] + (size_t)e * CD,
        (bhalf*)(ws + O_moeT + (size_t)(2 * 4 + e) * SZw), moe_b[2] + (size_t)e * CD,
        (bhalf*)(ws + O_moeT + (size_t)(3 * 4 + e) * SZw), moe_b[3] + (size_t)e * CD,
        EO, false);
    k_moe_accum<<<(unsigned)((eg + 255) / 256), 256, 0, stream>>>(EO, rgF, moeO, e);
  }

  // 3) sparse attention (top-K=64 threshold)
  mha((bhalf*)(ws + O_spT + 0 * SZw), sp_b[0], (bhalf*)(ws + O_spT + 1 * SZw), sp_b[1],
      (bhalf*)(ws + O_spT + 2 * SZw), sp_b[2], (bhalf*)(ws + O_spT + 3 * SZw), sp_b[3],
      spO, true);

  // 4) performer (linear attention)
  gemm(xb, (bhalf*)(ws + O_pfT + 0 * SZw), pf_b[0], nullptr, Qb, CT, CD, CD, CD, CD, CD,
       1, 1, 0, 0, 0, 0, 0, 0, 1.f, 0);
  gemm(xb, (bhalf*)(ws + O_pfT + 1 * SZw), pf_b[1], nullptr, Kb, CT, CD, CD, CD, CD, CD,
       1, 1, 0, 0, 0, 0, 0, 0, 1.f, 0);
  gemm(xb, (bhalf*)(ws + O_pfT + 2 * SZw), pf_b[2], nullptr, Vb, CT, CD, CD, CD, CD, CD,
       1, 1, 0, 0, 0, 0, 0, 0, 1.f, 0);
  {
    long nvt = (long)CBH * CDH * CS;
    k_transpose_bhead<<<(unsigned)((nvt + 255) / 256), 256, 0, stream>>>(Vb, Vt);
  }
  // pq/pk = relu(Q/K @ wf) per head
  gemm(Qb, wfT, nullptr, nullptr, pqB, CS, CMF, CDH, CD, CDH, CMF, CBH, CH,
       (long)CS * CD, (long)CDH, 0, 0, (long)CH * CS * CMF, (long)CS * CMF, 1.f, 2);
  gemm(Kb, wfT, nullptr, nullptr, pkB, CS, CMF, CDH, CD, CDH, CMF, CBH, CH,
       (long)CS * CD, (long)CDH, 0, 0, (long)CH * CS * CMF, (long)CS * CMF, 1.f, 2);
  {
    long n = (long)CBH * CS * CMF;
    k_transpose_b<<<(unsigned)((n + 255) / 256), 256, 0, stream>>>(pkB, pkTB, CS, CMF, n);
  }
  k_colsum<<<(CBH * CMF + 255) / 256, 256, 0, stream>>>(pkB, zF);
  // kv^T[d,m] = sum_s Vt[d,s] * pkT[m,s]
  gemm(Vt, pkTB, nullptr, nullptr, kvTB, CDH, CMF, CS, CS, CS, CMF, CBH, CBH,
       0, (long)CDH * CS, 0, (long)CS * CMF, 0, (long)CDH * CMF, 1.f, 0);
  // num[s,d] = sum_m pq[s,m] * kvT[d,m]  (merged f32)
  gemm(pqB, kvTB, nullptr, numF, nullptr, CS, CDH, CMF, CMF, CMF, CD, CBH, CH,
       (long)CH * CS * CMF, (long)CS * CMF, (long)CH * CDH * CMF, (long)CDH * CMF,
       (long)CS * CD, (long)CDH, 1.f, 0);
  k_pffin<<<CBH * CS / 8, 256, 0, stream>>>(numF, pqB, zF, AOb);
  gemm(AOb, (bhalf*)(ws + O_pfT + 3 * SZw), pf_b[3], pfO, nullptr, CT, CD, CD, CD, CD, CD,
       1, 1, 0, 0, 0, 0, 0, 0, 1.f, 0);

  // 5) combine + LN1
  k_combine_attn<<<(unsigned)((eg + 255) / 256), 256, 0, stream>>>(x, moeO, spO, pfO, tgF, agF, yF);
  k_layernorm<<<CT / 8, 256, 0, stream>>>(yF, n1_g, n1_b, x1F, x1b);

  // 6) FFN (gelu fused into GEMM1), gated residual, LN2, LN3
  gemm(x1b, f1T, ffn_b1, nullptr, h1B, CT, CFF, CD, CD, CD, CFF, 1, 1, 0, 0, 0, 0, 0, 0, 1.f, 1);
  gemm(h1B, f2T, ffn_b2, ffnF, nullptr, CT, CD, CFF, CFF, CFF, CD, 1, 1, 0, 0, 0, 0, 0, 0, 1.f, 0);
  k_combine_ffn<<<(unsigned)((eg + 255) / 256), 256, 0, stream>>>(x1F, ffnF, gwF, y2F);
  k_layernorm<<<CT / 8, 256, 0, stream>>>(y2F, n2_g, n2_b, x2F, nullptr);
  k_layernorm<<<CT / 8, 256, 0, stream>>>(x2F, n3_g, n3_b, out, nullptr);
}
